// MyRNN_42855183680056
// MI455X (gfx1250) — matrix operand order repaired and checked by reading, not compiled, not run on hardware
//
#include <hip/hip_runtime.h>
#include <cstdint>
#include <cstddef>

typedef __attribute__((ext_vector_type(16))) _Float16 v16h;
typedef __attribute__((ext_vector_type(8)))  float    v8f;

#define T_STEPS 8
#define BATCH   4
#define CIN     3
#define HEIGHT  30
#define WIDTH   32
#define HW      960           // HEIGHT*WIDTH
#define NTOT    3840          // BATCH*HW  (GEMM N)
#define HID     128
#define OC      512           // 4*HID     (GEMM M)
#define CPAD0   160           // pad 3+128 -> mult of 32
#define CPAD1   256           // 2*128

union FragAB {
  v16h     v;
  uint32_t u[8];
};

// ---- convert weights (OIHW f32) -> f16 [tap][oc][ci_pad] ---------------------
__global__ void convert_w_kernel(const float* __restrict__ W, _Float16* __restrict__ wf,
                                 int cin, int cpad) {
  int idx = blockIdx.x * 256 + threadIdx.x;
  int total = 9 * OC * cpad;
  if (idx >= total) return;
  int tap = idx / (OC * cpad);
  int rem = idx - tap * (OC * cpad);
  int oc  = rem / cpad;
  int ci  = rem - oc * cpad;
  float v = 0.f;
  if (ci < cin) {
    int ky = tap / 3, kx = tap - ky * 3;
    v = W[((size_t)(oc * cin + ci) * 3 + ky) * 3 + kx];
  }
  wf[idx] = (_Float16)v;
}

// ---- build xh[n][cpad] f16 = concat(x_t | prev-layer h_t , h_{t-1}) ----------
__global__ void pack_xh_kernel(_Float16* __restrict__ xh,
                               const float* __restrict__ x,        // layer0 input
                               const _Float16* __restrict__ seq_in,// layers>0
                               const float* __restrict__ h_state,
                               int t, int cpad, int is_l0) {
  int idx = (blockIdx.x * 256 + threadIdx.x) * 4;
  if (idx >= NTOT * cpad) return;
  int n  = idx / cpad;
  int c0 = idx - n * cpad;
#pragma unroll
  for (int u = 0; u < 4; ++u) {
    int c = c0 + u;
    float v;
    if (is_l0) {
      if (c < CIN) {
        int b = n / HW, s = n - b * HW;
        v = x[((size_t)(t * BATCH + b) * CIN + c) * HW + s];
      } else if (c < CIN + HID) {
        v = h_state[(size_t)n * HID + (c - CIN)];
      } else {
        v = 0.f;
      }
    } else {
      if (c < HID) v = (float)seq_in[((size_t)t * NTOT + n) * HID + c];
      else         v = h_state[(size_t)n * HID + (c - HID)];
    }
    xh[(size_t)n * cpad + c] = (_Float16)v;
  }
}

// ---- 3x3 conv as 9 shifted WMMA GEMMs with double-buffered LDS pipeline ------
// conv[n][oc] = sum_tap Wtap * shift_tap(xh)
__global__ __launch_bounds__(256)
void conv_gemm_kernel(const _Float16* __restrict__ wf,   // [9][OC][cpad]
                      const _Float16* __restrict__ xh,   // [NTOT][cpad]
                      float* __restrict__ convout,       // [NTOT][OC]
                      int cpad) {
  __shared__ _Float16 bsh[2][128][40];  // double buffer; row padded vs bank conflicts
  const int tid  = threadIdx.x;
  const int lane = tid & 31;
  const int wave = tid >> 5;
  const int wm   = wave & 3;                 // 4 waves tile M
  const int wn   = wave >> 2;                // 2 waves tile N
  const int m0   = blockIdx.y * 128 + wm * 32;
  const int n0   = blockIdx.x * 128;
  const int nw0  = n0 + wn * 64;
  const int l15  = lane & 15;
  const int kh   = lane >> 4;
  const int kcnt = cpad >> 5;
  const int total = 9 * kcnt;

  // hoisted per-thread staging geometry: 4 tasks = 128 rows x 8 chunks / 256 thr
  int t_nl[4], t_ch[4], t_row[4], t_y[4], t_x[4];
#pragma unroll
  for (int k = 0; k < 4; ++k) {
    int idx = tid + k * 256;
    int nl  = idx >> 3;
    int ch  = idx & 7;
    int n   = n0 + nl;
    int b   = n / HW;
    int s   = n - b * HW;
    t_nl[k] = nl; t_ch[k] = ch;
    t_row[k] = b * HW;
    t_y[k] = s >> 5;            // WIDTH == 32
    t_x[k] = s & 31;
  }

  auto load_regs = [&](int tap, int kc, uint2 tmp[4]) {
    const int dy = tap / 3 - 1;
    const int dx = tap % 3 - 1;
#pragma unroll
    for (int k = 0; k < 4; ++k) {
      int yy = t_y[k] + dy;
      int xx = t_x[k] + dx;
      uint32_t lo = 0u, hi = 0u;
      if ((unsigned)yy < (unsigned)HEIGHT && (unsigned)xx < (unsigned)WIDTH) {
        const uint32_t* p = (const uint32_t*)(xh +
            ((size_t)(t_row[k] + (yy << 5) + xx) * cpad + (kc << 5) + (t_ch[k] << 2)));
        lo = p[0];
        hi = p[1];
      }
      tmp[k].x = lo;
      tmp[k].y = hi;
    }
  };
  auto store_lds = [&](int buf, const uint2 tmp[4]) {
#pragma unroll
    for (int k = 0; k < 4; ++k) {
      uint32_t* d = (uint32_t*)&bsh[buf][t_nl[k]][t_ch[k] << 2];
      d[0] = tmp[k].x;
      d[1] = tmp[k].y;
    }
  };

  v8f acc[2][4] = {};

  // prologue: stage tile 0
  {
    uint2 t0[4];
    load_regs(0, 0, t0);
    store_lds(0, t0);
  }
  __syncthreads();

  int ctap = 0, ckc = 0;       // current tile
  int ntap = 0, nkc = 1;       // next tile
  if (nkc == kcnt) { nkc = 0; ntap = 1; }

  for (int it = 0; it < total; ++it) {
    const int cur = it & 1;
    const bool have_next = (it + 1 < total);

    // 1) issue next tile's global loads into registers (overlaps with WMMA)
    uint2 tmp[4];
    if (have_next) load_regs(ntap, nkc, tmp);

    // 2) compute current tile
    const _Float16* wtap = wf + (size_t)ctap * OC * cpad;
    FragAB a[2], bq[4];
#pragma unroll
    for (int ms = 0; ms < 2; ++ms) {
      // A 16x32 f16 per ISA layout: K = 2j + (j>=4?8:0) + 8*kh  -> two b128 loads
      const _Float16* wrow =
          wtap + (size_t)(m0 + ms * 16 + l15) * cpad + (ckc << 5) + kh * 8;
#pragma unroll
      for (int j = 0; j < 8; ++j)
        a[ms].u[j] = *(const uint32_t*)(wrow + 2 * j + ((j >= 4) ? 8 : 0));
    }
#pragma unroll
    for (int ns = 0; ns < 4; ++ns) {
      // B 32x16 f16 per ISA layout: lane&15 = N; K = 2j + (j>=4?8:0) + 8*kh
      const _Float16* brow = &bsh[cur][wn * 64 + ns * 16 + l15][kh * 8];
#pragma unroll
      for (int j = 0; j < 8; ++j)
        bq[ns].u[j] = *(const uint32_t*)(brow + 2 * j + ((j >= 4) ? 8 : 0));
    }
#pragma unroll
    for (int ms = 0; ms < 2; ++ms)
#pragma unroll
      for (int ns = 0; ns < 4; ++ns)
        acc[ms][ns] = __builtin_amdgcn_wmma_f32_16x16x32_f16(
            false, a[ms].v, false, bq[ns].v, (short)0, acc[ms][ns], false, false);

    // 3) commit staged tile into the other LDS buffer
    if (have_next) store_lds(cur ^ 1, tmp);

    // 4) single barrier: prior buffer consumed + new buffer visible
    __syncthreads();

    ctap = ntap; ckc = nkc;
    ++nkc;
    if (nkc == kcnt) { nkc = 0; ++ntap; }
  }

  // epilogue: C/D layout -> lane&15 = N, VGPR r -> M = r + 8*kh (32B contiguous)
#pragma unroll
  for (int ms = 0; ms < 2; ++ms)
#pragma unroll
    for (int ns = 0; ns < 4; ++ns) {
      int n = nw0 + ns * 16 + l15;
      int m = m0 + ms * 16 + kh * 8;
      *(v8f*)(convout + (size_t)n * OC + m) = acc[ms][ns];
    }
}

// ---- fused LSTM gates / state update ----------------------------------------
__global__ void lstm_kernel(const float* __restrict__ convout,
                            const float* __restrict__ bias,
                            float* __restrict__ h_state,
                            float* __restrict__ c_state,
                            _Float16* __restrict__ seq_out,
                            int t) {
  int idx = blockIdx.x * 256 + threadIdx.x;   // n*HID + j
  if (idx >= NTOT * HID) return;
  int n = idx >> 7;
  int j = idx & 127;
  const float* row = convout + (size_t)n * OC;
  float ai = row[j]       + bias[j];
  float af = row[128 + j] + bias[128 + j];
  float ao = row[256 + j] + bias[256 + j];
  float ag = row[384 + j] + bias[384 + j];
  float ig = 1.f / (1.f + __expf(-ai));
  float fg = 1.f / (1.f + __expf(-af));
  float og = 1.f / (1.f + __expf(-ao));
  float g  = tanhf(ag);
  float cn = fg * c_state[idx] + ig * g;
  float hn = og * tanhf(cn);
  c_state[idx] = cn;
  h_state[idx] = hn;
  seq_out[(size_t)t * (NTOT * HID) + idx] = (_Float16)hn;
}

// ---- head: z1 = relu(h_last @ lin1_w^T + b1) --------------------------------
__global__ void head1_kernel(const float* __restrict__ h_state,  // [n][HID]
                             const float* __restrict__ w1,       // [256][HID*HW]
                             const float* __restrict__ b1,
                             float* __restrict__ z1) {
  __shared__ float red[256];
  int b   = blockIdx.x >> 8;
  int o   = blockIdx.x & 255;
  int tid = threadIdx.x;
  const float* wrow = w1 + (size_t)o * (HID * HW);
  float sum = 0.f;
  for (int c = 0; c < HID; ++c) {
    const float* wc = wrow + c * HW;           // ref flat index = c*H*W + s
    for (int s = tid; s < HW; s += 256)
      sum += wc[s] * h_state[(size_t)(b * HW + s) * HID + c];
  }
  red[tid] = sum;
  __syncthreads();
  for (int st = 128; st > 0; st >>= 1) {
    if (tid < st) red[tid] += red[tid + st];
    __syncthreads();
  }
  if (tid == 0) {
    float v = red[0] + b1[o];
    z1[b * 256 + o] = v > 0.f ? v : 0.f;
  }
}

__global__ void head2_kernel(const float* __restrict__ z1,
                             const float* __restrict__ w2,  // [2][256]
                             const float* __restrict__ b2,
                             float* __restrict__ out) {     // mu(4) then sigma(4)
  int tid = threadIdx.x;
  if (tid < 8) {
    int b = tid & 3;
    int k = tid >> 2;
    float sum = b2[k];
    for (int o = 0; o < 256; ++o) sum += w2[k * 256 + o] * z1[b * 256 + o];
    out[k * 4 + b] = sum;
  }
}

extern "C" void kernel_launch(void* const* d_in, const int* in_sizes, int n_in,
                              void* d_out, int out_size, void* d_ws, size_t ws_size,
                              hipStream_t stream) {
  const float* x      = (const float*)d_in[0];
  const float* W[4]   = {(const float*)d_in[1], (const float*)d_in[3],
                         (const float*)d_in[5], (const float*)d_in[7]};
  const float* bv[4]  = {(const float*)d_in[2], (const float*)d_in[4],
                         (const float*)d_in[6], (const float*)d_in[8]};
  const float* lin1_w = (const float*)d_in[9];
  const float* lin1_b = (const float*)d_in[10];
  const float* lin2_w = (const float*)d_in[11];
  const float* lin2_b = (const float*)d_in[12];
  float* out = (float*)d_out;

  // workspace carve-out (~38 MB total)
  char*  base = (char*)d_ws;
  size_t off  = 0;
  auto take = [&](size_t bytes) -> char* {
    char* p = base + off;
    off = (off + bytes + 255) & ~(size_t)255;
    return p;
  };
  _Float16* wf[4];
  wf[0] = (_Float16*)take((size_t)9 * OC * CPAD0 * 2);
  for (int l = 1; l < 4; ++l) wf[l] = (_Float16*)take((size_t)9 * OC * CPAD1 * 2);
  _Float16* xh     = (_Float16*)take((size_t)NTOT * CPAD1 * 2);
  float*    convb  = (float*)take((size_t)NTOT * OC * 4);
  float*    hstate = (float*)take((size_t)NTOT * HID * 4);
  float*    cstate = (float*)take((size_t)NTOT * HID * 4);
  _Float16* seq[2];
  seq[0] = (_Float16*)take((size_t)T_STEPS * NTOT * HID * 2);
  seq[1] = (_Float16*)take((size_t)T_STEPS * NTOT * HID * 2);
  float* z1 = (float*)take(4 * 256 * 4);

  // one-time (per launch) weight f16 repack — deterministic
  {
    int tot0 = 9 * OC * CPAD0;
    convert_w_kernel<<<(tot0 + 255) / 256, 256, 0, stream>>>(W[0], wf[0], CIN + HID, CPAD0);
    int tot1 = 9 * OC * CPAD1;
    for (int l = 1; l < 4; ++l)
      convert_w_kernel<<<(tot1 + 255) / 256, 256, 0, stream>>>(W[l], wf[l], 2 * HID, CPAD1);
  }

  for (int layer = 0; layer < 4; ++layer) {
    int cpad = (layer == 0) ? CPAD0 : CPAD1;
    const _Float16* sin  = (layer == 0) ? nullptr : seq[(layer & 1) ^ 1];
    _Float16*       sout = seq[layer & 1];
    hipMemsetAsync(hstate, 0, (size_t)NTOT * HID * 4, stream);
    hipMemsetAsync(cstate, 0, (size_t)NTOT * HID * 4, stream);
    for (int t = 0; t < T_STEPS; ++t) {
      int packtot = NTOT * cpad / 4;
      pack_xh_kernel<<<(packtot + 255) / 256, 256, 0, stream>>>(
          xh, x, sin, hstate, t, cpad, layer == 0 ? 1 : 0);
      conv_gemm_kernel<<<dim3(NTOT / 128, OC / 128), 256, 0, stream>>>(
          wf[layer], xh, convb, cpad);
      lstm_kernel<<<(NTOT * HID + 255) / 256, 256, 0, stream>>>(
          convb, bv[layer], hstate, cstate, sout, t);
    }
  }
  head1_kernel<<<BATCH * 256, 256, 0, stream>>>(hstate, lin1_w, lin1_b, z1);
  head2_kernel<<<1, 32, 0, stream>>>(z1, lin2_w, lin2_b, out);
}
